// DnlsLoss_18571438588509
// MI455X (gfx1250) — matrix operand order repaired and checked by reading, not compiled, not run on hardware
//
#include <hip/hip_runtime.h>

typedef __attribute__((ext_vector_type(16))) _Float16 v16h;
typedef __attribute__((ext_vector_type(8)))  _Float16 v8h;
typedef __attribute__((ext_vector_type(8)))  float    v8f;

#define Tn   5
#define Cn   3
#define Hn   256
#define Wn   256
#define PSn  7
#define PELEM (PSn*PSn*Cn)   // 147
#define KPAD  224            // 7 patch rows * 32 (row = 21 elems + 11 zero pad)
#define NHn  64
#define NWn  64
#define QTOT (Tn*NHn*NWn)    // 20480
#define NCAND 243            // 3 * 81

__device__ __forceinline__ int iclip(int v, int lo, int hi) {
    return v < lo ? lo : (v > hi ? hi : v);
}

__global__ void dnls_init_out(float* out) { out[0] = 0.0f; }

// One wave32 workgroup per tile of 16 queries sharing (t, hq-row, 16 lattice columns).
__global__ __launch_bounds__(32) void dnls_kernel(
    const float* __restrict__ noisy,
    const float* __restrict__ deno,
    const float* __restrict__ fflow,
    const float* __restrict__ bflow,
    const int*   __restrict__ epoch,
    float*       __restrict__ out)
{
    // self-frame pixel region covering all block-0 candidates + patches of this tile
    __shared__ float    region[15 * 75 * 3];  // rows hq-4..hq+10, cols w0-4..w0+70, ch
    __shared__ float    Aq[16][PELEM];        // f32 query patches (noisy), e=(ph*7+pw)*3+c
    __shared__ _Float16 Ah[16][KPAD];         // f16 query patches, row-chunk K order
    __shared__ float    qn[16];               // query squared norms
    __shared__ float    Dall[16][244];        // all 243 candidate distances
    __shared__ float    Cmat[16][80];         // q·c dot products for one dh row
    __shared__ float    cnormRow[80];         // candidate norms for one dh row
    __shared__ int      ffx[16], ffy[16], bfx[16], bfy[16];
    __shared__ int      nbr[16][10];

    const int tile = blockIdx.x;
    const int tw   = tile & 3;          // 4 tiles of 16 queries per row (nW=64)
    const int hqi  = (tile >> 2) & 63;
    const int t    = tile >> 8;
    const int hq   = hqi * 4;
    const int w0   = tw * 64;           // wq_j = w0 + 4*j
    const int L    = threadIdx.x;
    const int Nc   = L & 15;
    const int khalf = L >> 4;

    const int ep   = epoch[0];
    int keff = (int)(10.0 * ((double)(100 - ep) / 100.0));
    keff = keff < 2 ? 2 : (keff > 10 ? 10 : keff);

    // ---- stage self-frame region in LDS (clipped at image borders) ----
    const float* frame_t = noisy + (size_t)t * Cn * Hn * Wn;
    for (int idx = L; idx < 15 * 75 * 3; idx += 32) {
        int rr = idx / (75 * 3); int r2 = idx % (75 * 3);
        int cc = r2 / 3;         int ch = r2 % 3;
        int ir = iclip(hq - 4 + rr, 0, Hn - 1);
        int ic = iclip(w0 - 4 + cc, 0, Wn - 1);
        region[idx] = frame_t[((size_t)ch * Hn + ir) * Wn + ic];
    }
    __syncthreads();

    // ---- stage f32 query patches from the region ----
    for (int idx = L; idx < 16 * PELEM; idx += 32) {
        int j = idx / PELEM; int e = idx % PELEM;
        int ph = e / 21; int q = e % 21; int pw = q / 3; int ch = q % 3;
        int rr = (hq + ph > Hn - 1 ? Hn - 1 : hq + ph) - (hq - 4);
        int wp = w0 + 4 * j + pw;
        int cc = (wp > Wn - 1 ? Wn - 1 : wp) - (w0 - 4);
        Aq[j][e] = region[(rr * 75 + cc) * 3 + ch];
    }
    __syncthreads();

    // ---- f16 copy in WMMA K order: k = ph*32 + q (q = pw*3+c, zero-padded q>=21) ----
    for (int idx = L; idx < 16 * KPAD; idx += 32) {
        int j = idx / KPAD; int k = idx % KPAD;
        int ph = k >> 5;    int q = k & 31;
        float v = (q < 21) ? Aq[j][ph * 21 + q] : 0.0f;
        Ah[j][k] = (_Float16)v;
    }
    __syncthreads();

    // ---- query norms + rounded flows (round-half-even, matching jnp.round) ----
    if (L < 16) {
        float s = 0.0f;
        for (int e = 0; e < PELEM; ++e) { float v = Aq[L][e]; s += v * v; }
        qn[L] = s;
        int wq = w0 + 4 * L;
        size_t fb  = (((size_t)t * 2 + 0) * Hn + hq) * Wn + wq;
        size_t fb1 = (((size_t)t * 2 + 1) * Hn + hq) * Wn + wq;
        ffx[L] = (int)rintf(fflow[fb]);  ffy[L] = (int)rintf(fflow[fb1]);
        bfx[L] = (int)rintf(bflow[fb]);  bfy[L] = (int)rintf(bflow[fb1]);
    }
    __syncthreads();

    // ---- hoist A operand into registers (tile-invariant across dh/ct) ----
    // A operand order used here (16x32 f16): per lane two 8xf16 runs at kl = off0 and off0+8 (contiguous k = 16*khalf + s)
    v16h areg[7];
    {
        const _Float16* AhRow = &Ah[Nc][0];
        const int off0 = khalf ? 16 : 0;
        #pragma unroll
        for (int kc = 0; kc < 7; ++kc) {
            v8h lo = *(const v8h*)(AhRow + 32 * kc + off0);
            v8h hi = *(const v8h*)(AhRow + 32 * kc + off0 + 8);
            #pragma unroll
            for (int i = 0; i < 8; ++i) { areg[kc][i] = lo[i]; areg[kc][i + 8] = hi[i]; }
        }
    }

    // ================= block 0 (self frame): shared-column WMMA GEMM =================
    // D[j, dh, dw] = ||q_j||^2 + ||c||^2 - 2 q_j·c via v_wmma_f32_16x16x32_f16.
    // 16 queries (M) x shared candidate columns (N), K chunk kc == patch row ph.
    for (int dhi = 0; dhi < 9; ++dhi) {
        const int hcand = iclip(hq + dhi - 4, 0, Hn - 1);

        for (int ct = 0; ct < 5; ++ct) {
            int g = 16 * ct + Nc; if (g > 68) g = 68;   // pad columns duplicate col 68
            const int wcand = iclip(w0 - 4 + g, 0, Wn - 1);
            int ccL[7];                                  // compile-time indexed only
            #pragma unroll
            for (int pw = 0; pw < PSn; ++pw) {
                int wp = wcand + pw;
                ccL[pw] = (wp > Wn - 1 ? Wn - 1 : wp) - (w0 - 4);
            }

            v8f acc = {};
            float pn = 0.0f;                             // partial ||c||^2 (this K-half)
            #pragma unroll
            for (int kc = 0; kc < 7; ++kc) {
                int hp = hcand + kc;
                int rr = (hp > Hn - 1 ? Hn - 1 : hp) - (hq - 4);
                const float* row = &region[rr * 225];

                v16h b = {};
                if (khalf == 0) {                        // q = 0..15, all valid
                    #pragma unroll
                    for (int i = 0; i < 16; ++i) {
                        float fv = row[ccL[i / 3] * 3 + (i % 3)];
                        b[i] = (_Float16)fv;
                        pn += fv * fv;
                    }
                } else {                                 // q = 16..20 valid, rest zero pad
                    #pragma unroll
                    for (int i = 0; i < 5; ++i) {
                        const int q = 16 + i;
                        float fv = row[ccL[q / 3] * 3 + (q % 3)];
                        b[i] = (_Float16)fv;
                        pn += fv * fv;
                    }
                }
                acc = __builtin_amdgcn_wmma_f32_16x16x32_f16(
                    false, areg[kc], false, b, (short)0, acc, false, false);
            }
            // merge the two K-halves of the column norm (lane L <-> L+16 share column N)
            float full = pn + __shfl_xor(pn, 16, 32);
            if (L < 16) cnormRow[16 * ct + Nc] = full;
            // scatter D-tile: element (M = r + 8*khalf, N = Nc)
            #pragma unroll
            for (int r = 0; r < 8; ++r)
                Cmat[r + 8 * khalf][16 * ct + Nc] = acc[r];
        }
        __syncthreads();

        // assemble distances for this dh: query j at dw reads shared column g = 4j + dwi
        for (int ii = L; ii < 144; ii += 32) {
            int j = ii & 15; int dwi = ii >> 4;
            int gq = 4 * j + dwi;
            Dall[j][dhi * 9 + dwi] = qn[j] + cnormRow[gq] - 2.0f * Cmat[j][gq];
        }
        __syncthreads();
    }

    // ================= blocks 1,2 (flow-warped prev/next): per-query VALU SSD =========
    {
        int blk = khalf;                          // 0 -> backward (t-1), 1 -> forward (t+1)
        int j   = Nc;
        int tc  = blk ? iclip(t + 1, 0, Tn - 1) : iclip(t - 1, 0, Tn - 1);
        int hcen = iclip(hq + (blk ? ffy[j] : bfy[j]), 0, Hn - 1);
        int wcen = iclip(w0 + 4 * j + (blk ? ffx[j] : bfx[j]), 0, Wn - 1);
        const float* fr = noisy + (size_t)tc * Cn * Hn * Wn;
        for (int o = 0; o < 81; ++o) {
            int hc = iclip(hcen + o / 9 - 4, 0, Hn - 1);
            int wc = iclip(wcen + o % 9 - 4, 0, Wn - 1);
            float s = 0.0f; int e = 0;
            for (int ph = 0; ph < PSn; ++ph) {
                int ir = hc + ph > Hn - 1 ? Hn - 1 : hc + ph;
                for (int pw = 0; pw < PSn; ++pw) {
                    int ic = wc + pw > Wn - 1 ? Wn - 1 : wc + pw;
                    #pragma unroll
                    for (int ch = 0; ch < Cn; ++ch) {
                        float cv = fr[((size_t)ch * Hn + ir) * Wn + ic];
                        float d = Aq[j][e++] - cv;
                        s += d * d;
                    }
                }
            }
            Dall[j][81 + blk * 81 + o] = s;
        }
    }
    __syncthreads();

    // ================= anchored top-k (self candidate n=40 forced to slot 0) ==========
    if (L < 16) {
        nbr[L][0] = 40;
        Dall[L][40] = __builtin_inff();
        for (int k = 1; k < keff; ++k) {
            float best = __builtin_inff(); int bi = 0;
            for (int n = 0; n < NCAND; ++n) {
                float d = Dall[L][n];
                if (d < best) { best = d; bi = n; }   // strict <: lowest index on ties
            }
            nbr[L][k] = bi;
            Dall[L][bi] = __builtin_inff();
        }
    }
    __syncthreads();

    // ================= refine: SSD(deno query patch, noisy neighbor patch), k=1.. =====
    float accum = 0.0f;
    const int tasks = 16 * (keff - 1);
    for (int ii = L; ii < tasks; ii += 32) {
        int j = ii / (keff - 1); int kk = 1 + ii % (keff - 1);
        int n = nbr[j][kk];
        int dti = n / 81; int o = n % 81;
        int dh = o / 9 - 4, dw = o % 9 - 4;
        int wq = w0 + 4 * j;
        int tc, hcen, wcen;
        if (dti == 0)      { tc = t;                     hcen = hq;                             wcen = wq; }
        else if (dti == 1) { tc = iclip(t - 1, 0, Tn-1); hcen = iclip(hq + bfy[j], 0, Hn - 1);  wcen = iclip(wq + bfx[j], 0, Wn - 1); }
        else               { tc = iclip(t + 1, 0, Tn-1); hcen = iclip(hq + ffy[j], 0, Hn - 1);  wcen = iclip(wq + ffx[j], 0, Wn - 1); }
        int hc = iclip(hcen + dh, 0, Hn - 1);
        int wc = iclip(wcen + dw, 0, Wn - 1);
        const float* dfr = deno  + (size_t)t  * Cn * Hn * Wn;
        const float* nfr = noisy + (size_t)tc * Cn * Hn * Wn;
        float s = 0.0f;
        for (int ph = 0; ph < PSn; ++ph) {
            int irq = hq + ph > Hn - 1 ? Hn - 1 : hq + ph;
            int irk = hc + ph > Hn - 1 ? Hn - 1 : hc + ph;
            for (int pw = 0; pw < PSn; ++pw) {
                int icq = wq + pw > Wn - 1 ? Wn - 1 : wq + pw;
                int ick = wc + pw > Wn - 1 ? Wn - 1 : wc + pw;
                #pragma unroll
                for (int ch = 0; ch < Cn; ++ch) {
                    float qv = dfr[((size_t)ch * Hn + irq) * Wn + icq];
                    float kv = nfr[((size_t)ch * Hn + irk) * Wn + ick];
                    float d = qv - kv;
                    s += d * d;
                }
            }
        }
        accum += s;
    }
    // wave32 reduction + single atomic per tile
    for (int off = 16; off > 0; off >>= 1)
        accum += __shfl_xor(accum, off, 32);
    if (L == 0) {
        float denom = (float)QTOT * (float)(keff - 1);
        atomicAdd(out, accum / denom);
    }
}

extern "C" void kernel_launch(void* const* d_in, const int* in_sizes, int n_in,
                              void* d_out, int out_size, void* d_ws, size_t ws_size,
                              hipStream_t stream) {
    const float* noisy = (const float*)d_in[0];
    const float* deno  = (const float*)d_in[1];
    const float* fflow = (const float*)d_in[2];
    const float* bflow = (const float*)d_in[3];
    const int*   epoch = (const int*)d_in[4];
    float* out = (float*)d_out;

    dnls_init_out<<<1, 1, 0, stream>>>(out);
    // 5 frames * 64 rows * 4 column-tiles = 1280 tiles, one wave32 each
    dnls_kernel<<<Tn * NHn * (NWn / 16), 32, 0, stream>>>(
        noisy, deno, fflow, bflow, epoch, out);
}
